// LSTM_AE_86981677679220
// MI455X (gfx1250) — compile-verified, matrix operand order checked by reading, not run on hardware
//
#include <hip/hip_runtime.h>

// ---------- types ----------
typedef __attribute__((ext_vector_type(16))) __bf16        v16bf;
typedef __attribute__((ext_vector_type(8)))  float         v8f;
typedef __attribute__((ext_vector_type(4)))  unsigned int  v4u;

union Frag16 { v16bf v; v4u q[2]; };

__device__ __forceinline__ unsigned short f2bf(float f) {
  unsigned int u = __float_as_uint(f);
  unsigned int r = (u + 0x7FFFu + ((u >> 16) & 1u)) >> 16;   // RNE
  return (unsigned short)r;
}

__device__ __forceinline__ float sigf(float x) { return 1.0f / (1.0f + __expf(-x)); }

#define WMMA_BF16(A, B, C) \
  __builtin_amdgcn_wmma_f32_16x16x32_bf16(false, (A), false, (B), (short)0, (C), false, false)

// ---------- small utility kernels ----------
__global__ void k_f32_to_bf16(const float* __restrict__ in,
                              unsigned short* __restrict__ out, int n) {
  int i = blockIdx.x * blockDim.x + threadIdx.x;
  if (i < n) out[i] = f2bf(in[i]);
}

__global__ void k_bias_combine(const float* __restrict__ a,
                               const float* __restrict__ b,
                               float* __restrict__ out, int n) {
  int i = blockIdx.x * blockDim.x + threadIdx.x;
  if (i < n) out[i] = a[i] + b[i];
}

__global__ void k_copy_f32(const float* __restrict__ in, float* __restrict__ out, int n) {
  int i = blockIdx.x * blockDim.x + threadIdx.x;
  if (i < n) out[i] = in[i];
}

__global__ void k_fill_f32(float* __restrict__ p, float v, int n) {
  int i = blockIdx.x * blockDim.x + threadIdx.x;
  if (i < n) p[i] = v;
}

__global__ void k_fill_bf16(unsigned short* __restrict__ p, unsigned short v, int n) {
  int i = blockIdx.x * blockDim.x + threadIdx.x;
  if (i < n) p[i] = v;
}

// dec_in[b,0,:] = emb[b,:];  dec_in[b,t,:] = x[b,t-1, 1023-j]  (feature flip)
__global__ void k_build_dec_in(const float* __restrict__ x,
                               const float* __restrict__ emb,
                               float* __restrict__ dec_in) {
  int idx = blockIdx.x * blockDim.x + threadIdx.x;     // 64*256*1024
  int j = idx & 1023;
  int t = (idx >> 10) & 255;
  int b = idx >> 18;
  float v;
  if (t == 0) v = emb[(b << 10) + j];
  else        v = x[(((size_t)b * 256 + (t - 1)) << 10) + (1023 - j)];
  dec_in[idx] = v;
}

// ---------- big WMMA GEMM:  C[m,n] = sum_k A[m,k]*W[n,k] ----------
// A: bf16 [M,K] row-major.  W: bf16 [N,K] row-major (B = W^T).
// One wave -> 16(M) x 64(N) strip: 1 A-frag reused over 4 B-frags per K-step.
__global__ __launch_bounds__(256)
void k_gemm_wmma(const unsigned short* __restrict__ A,
                 const unsigned short* __restrict__ W,
                 float* __restrict__ C,
                 int M, int N, int K)
{
  int gw = blockIdx.x * (blockDim.x >> 5) + (threadIdx.x >> 5);
  int tilesM = M >> 4;
  int tilesN = N >> 6;
  if (gw >= tilesM * tilesN) return;                 // wave-uniform: EXEC stays all-ones
  int tm = gw % tilesM;
  int tn = gw / tilesM;
  int m0 = tm << 4;
  int n0 = tn << 6;
  int lane = threadIdx.x & 31;
  int ml = lane & 15;
  int hi = lane >> 4;

  const unsigned short* aRow = A + (size_t)(m0 + ml) * K + hi * 8;
  const unsigned short* b0   = W + (size_t)(n0 + ml) * K + hi * 8;
  const unsigned short* b1   = b0 + (size_t)16 * K;
  const unsigned short* b2   = b0 + (size_t)32 * K;
  const unsigned short* b3   = b0 + (size_t)48 * K;

  v8f acc0 = {}, acc1 = {}, acc2 = {}, acc3 = {};
  for (int k = 0; k < K; k += 32) {
    Frag16 fa, f0, f1, f2, f3;
    fa.q[0] = *(const v4u*)(aRow + k); fa.q[1] = *(const v4u*)(aRow + k + 16);
    f0.q[0] = *(const v4u*)(b0 + k);   f0.q[1] = *(const v4u*)(b0 + k + 16);
    f1.q[0] = *(const v4u*)(b1 + k);   f1.q[1] = *(const v4u*)(b1 + k + 16);
    f2.q[0] = *(const v4u*)(b2 + k);   f2.q[1] = *(const v4u*)(b2 + k + 16);
    f3.q[0] = *(const v4u*)(b3 + k);   f3.q[1] = *(const v4u*)(b3 + k + 16);
    acc0 = WMMA_BF16(fa.v, f0.v, acc0);
    acc1 = WMMA_BF16(fa.v, f1.v, acc1);
    acc2 = WMMA_BF16(fa.v, f2.v, acc2);
    acc3 = WMMA_BF16(fa.v, f3.v, acc3);
  }

  int col = n0 + ml;                                  // D layout: N = lane&15
#pragma unroll
  for (int r = 0; r < 8; ++r) {
    int row = m0 + r + 8 * hi;                        // D layout: M = r + 8*(lane>>4)
    float* cr = C + (size_t)row * N;
    cr[col]      = acc0[r];
    cr[col + 16] = acc1[r];
    cr[col + 32] = acc2[r];
    cr[col + 48] = acc3[r];
  }
}

// ---------- fused recurrent step: gates = G[t] + h@Whh^T + bias; cell update ----------
// Block = 128 threads (4 waves). Block owns one 16(batch) x 16(hidden) patch;
// each wave accumulates all 4 gate tiles over K/4, LDS-reduce, fused LSTM epilogue.
__global__ __launch_bounds__(128)
void k_lstm_step(const unsigned short* __restrict__ hbfIn,  // [64,1024] bf16
                 const unsigned short* __restrict__ Whh,    // [4096,1024] bf16
                 const float* __restrict__ Gt,              // G + t*4096, row stride ldG
                 const float* __restrict__ bias,            // [4096]
                 float* __restrict__ h, float* __restrict__ c,   // [64,1024] f32
                 unsigned short* __restrict__ hbfOut,       // [64,1024] bf16 (ping-pong)
                 float* __restrict__ outSeq,                // nullable
                 long ldG, int t, int flip, int S)
{
  constexpr int K = 1024;
  int tm = blockIdx.x & 3;          // batch tile (M=64 -> 4 tiles)
  int tn = blockIdx.x >> 2;         // hidden tile (1024 -> 64 tiles)
  int m0 = tm << 4;
  int n0 = tn << 4;
  int wave = threadIdx.x >> 5;      // 0..3 : K-split
  int lane = threadIdx.x & 31;
  int ml = lane & 15;
  int hi = lane >> 4;

  const unsigned short* aRow = hbfIn + (size_t)(m0 + ml) * K + hi * 8 + wave * 256;
  const unsigned short* bI = Whh + (size_t)(n0 + ml) * K + hi * 8 + wave * 256;
  const unsigned short* bF = bI + (size_t)1024 * K;
  const unsigned short* bG = bI + (size_t)2048 * K;
  const unsigned short* bO = bI + (size_t)3072 * K;

  v8f aI = {}, aF = {}, aG = {}, aO = {};
#pragma unroll
  for (int k = 0; k < 256; k += 32) {                 // 8 fully-unrolled k-steps
    Frag16 fa, f0, f1, f2, f3;
    fa.q[0] = *(const v4u*)(aRow + k); fa.q[1] = *(const v4u*)(aRow + k + 16);
    f0.q[0] = *(const v4u*)(bI + k);   f0.q[1] = *(const v4u*)(bI + k + 16);
    f1.q[0] = *(const v4u*)(bF + k);   f1.q[1] = *(const v4u*)(bF + k + 16);
    f2.q[0] = *(const v4u*)(bG + k);   f2.q[1] = *(const v4u*)(bG + k + 16);
    f3.q[0] = *(const v4u*)(bO + k);   f3.q[1] = *(const v4u*)(bO + k + 16);
    aI = WMMA_BF16(fa.v, f0.v, aI);
    aF = WMMA_BF16(fa.v, f1.v, aF);
    aG = WMMA_BF16(fa.v, f2.v, aG);
    aO = WMMA_BF16(fa.v, f3.v, aO);
  }

  // K-split reduction through LDS (pad to 33 to avoid 32-way bank conflicts)
  __shared__ float red[4][32][33];
  float* dst = &red[wave][lane][0];
#pragma unroll
  for (int r = 0; r < 8; ++r) {
    dst[r]      = aI[r];
    dst[8 + r]  = aF[r];
    dst[16 + r] = aG[r];
    dst[24 + r] = aO[r];
  }
  __syncthreads();

  // Each wave finishes 2 of the 8 accumulator rows (all waves busy in epilogue)
#pragma unroll
  for (int rl = 0; rl < 2; ++rl) {
    int r = (wave << 1) + rl;
    float vI = red[0][lane][r]      + red[1][lane][r]      + red[2][lane][r]      + red[3][lane][r];
    float vF = red[0][lane][8 + r]  + red[1][lane][8 + r]  + red[2][lane][8 + r]  + red[3][lane][8 + r];
    float vG = red[0][lane][16 + r] + red[1][lane][16 + r] + red[2][lane][16 + r] + red[3][lane][16 + r];
    float vO = red[0][lane][24 + r] + red[1][lane][24 + r] + red[2][lane][24 + r] + red[3][lane][24 + r];

    int row  = m0 + r + 8 * hi;     // batch index (D layout: M = r + 8*(lane>>4))
    int colh = n0 + ml;             // hidden index (D layout: N = lane&15)
    const float* gr = Gt + (size_t)row * ldG;
    vI += gr[colh]        + bias[colh];
    vF += gr[1024 + colh] + bias[1024 + colh];
    vG += gr[2048 + colh] + bias[2048 + colh];
    vO += gr[3072 + colh] + bias[3072 + colh];

    int idx = (row << 10) + colh;
    float cn = sigf(vF) * c[idx] + sigf(vI) * tanhf(vG);
    float hn = sigf(vO) * tanhf(cn);
    c[idx] = cn;
    h[idx] = hn;
    hbfOut[idx] = f2bf(hn);
    if (outSeq) {
      int jj = flip ? (1023 - colh) : colh;
      outSeq[((size_t)row * S + t) * 1024 + jj] = hn;
    }
  }
}

// ---------- host orchestration ----------
extern "C" void kernel_launch(void* const* d_in, const int* in_sizes, int n_in,
                              void* d_out, int out_size, void* d_ws, size_t ws_size,
                              hipStream_t stream)
{
  (void)in_sizes; (void)n_in; (void)out_size; (void)ws_size;
  constexpr int Bm = 64, Sm = 256, Hm = 1024, G4 = 4096;
  const size_t WSZ = (size_t)G4 * Hm;                 // 4M elems per weight matrix
  const int seqElems = Bm * Sm * Hm;                  // 16,777,216
  const long ldStep = (long)Sm * G4;

  const float* x = (const float*)d_in[0];
  const float* Wih[4] = {(const float*)d_in[1], (const float*)d_in[5],
                         (const float*)d_in[9], (const float*)d_in[13]};
  const float* Whh[4] = {(const float*)d_in[2], (const float*)d_in[6],
                         (const float*)d_in[10], (const float*)d_in[14]};
  const float* bih[4] = {(const float*)d_in[3], (const float*)d_in[7],
                         (const float*)d_in[11], (const float*)d_in[15]};
  const float* bhh[4] = {(const float*)d_in[4], (const float*)d_in[8],
                         (const float*)d_in[12], (const float*)d_in[16]};

  // workspace carve-up (~482 MB)
  char* p = (char*)d_ws;
  unsigned short *wihB[4], *whhB[4];
  for (int l = 0; l < 4; ++l) {
    wihB[l] = (unsigned short*)p; p += WSZ * 2;
    whhB[l] = (unsigned short*)p; p += WSZ * 2;
  }
  float* bias[4];
  for (int l = 0; l < 4; ++l) { bias[l] = (float*)p; p += (size_t)G4 * 4; }
  float* G    = (float*)p; p += (size_t)Bm * Sm * G4 * 4;   // 256 MB
  float* seqA = (float*)p; p += (size_t)seqElems * 4;       // 64 MB
  float* seqB = (float*)p; p += (size_t)seqElems * 4;       // 64 MB
  unsigned short* Abf = (unsigned short*)p; p += (size_t)seqElems * 2;  // 32 MB
  float* h0 = (float*)p; p += (size_t)Bm * Hm * 4;
  float* c0 = (float*)p; p += (size_t)Bm * Hm * 4;
  float* h1 = (float*)p; p += (size_t)Bm * Hm * 4;
  float* c1 = (float*)p; p += (size_t)Bm * Hm * 4;
  unsigned short* hpp[2];
  hpp[0] = (unsigned short*)p; p += (size_t)Bm * Hm * 2;
  hpp[1] = (unsigned short*)p; p += (size_t)Bm * Hm * 2;

  auto cdiv = [](int a, int b) { return (a + b - 1) / b; };

  // 1) weights -> bf16, combined biases
  for (int l = 0; l < 4; ++l) {
    k_f32_to_bf16 <<<cdiv((int)WSZ, 256), 256, 0, stream>>>(Wih[l], wihB[l], (int)WSZ);
    k_f32_to_bf16 <<<cdiv((int)WSZ, 256), 256, 0, stream>>>(Whh[l], whhB[l], (int)WSZ);
    k_bias_combine<<<cdiv(G4, 256), 256, 0, stream>>>(bih[l], bhh[l], bias[l], G4);
  }
  // 2) zero encoder states (h0,c0,h1,c1 are contiguous)
  k_fill_f32<<<cdiv(4 * Bm * Hm, 256), 256, 0, stream>>>(h0, 0.0f, 4 * Bm * Hm);

  auto run_layer = [&](const float* inSeqF, int wl, float* hs, float* cs,
                       float* outSeq, int flip, bool initHbfFromH) {
    // precompute G = inSeq @ Wih^T for all timesteps (big WMMA GEMM)
    k_f32_to_bf16<<<cdiv(seqElems, 256), 256, 0, stream>>>(inSeqF, Abf, seqElems);
    {
      int M = Bm * Sm, N = G4, K = Hm;
      int tiles = (M >> 4) * (N >> 6);
      k_gemm_wmma<<<cdiv(tiles, 8), 256, 0, stream>>>(Abf, wihB[wl], G, M, N, K);
    }
    if (initHbfFromH)
      k_f32_to_bf16<<<cdiv(Bm * Hm, 256), 256, 0, stream>>>(hs, hpp[0], Bm * Hm);
    else
      k_fill_bf16<<<cdiv(Bm * Hm, 256), 256, 0, stream>>>(hpp[0], (unsigned short)0, Bm * Hm);
    // sequential recurrence, fully fused per step; ping-pong bf16 h across steps
    for (int t = 0; t < Sm; ++t) {
      k_lstm_step<<<(Bm >> 4) * (Hm >> 4), 128, 0, stream>>>(
          hpp[t & 1], whhB[wl], G + (size_t)t * G4, bias[wl],
          hs, cs, hpp[(t + 1) & 1], outSeq, ldStep, t, flip, Sm);
    }
  };

  float* out_emb = (float*)d_out;
  float* out_dec = (float*)d_out + (size_t)Bm * Hm;

  run_layer(x,    0, h0, c0, seqA,    0, false);  // encoder layer 0 -> ys0
  run_layer(seqA, 1, h1, c1, nullptr, 0, false);  // encoder layer 1 (only h_n,c_n needed)
  k_copy_f32<<<cdiv(Bm * Hm, 256), 256, 0, stream>>>(h1, out_emb, Bm * Hm);  // emb
  k_build_dec_in<<<cdiv(seqElems, 256), 256, 0, stream>>>(x, h1, seqA);      // dec_in
  run_layer(seqA, 2, h0, c0, seqB,    0, true);   // decoder layer 0 (init = enc h_n[0])
  run_layer(seqB, 3, h1, c1, out_dec, 1, true);   // decoder layer 1 -> flipped decoded
}
